// MoE_layer_megatron_wo_gate_v2_46712064311450
// MI455X (gfx1250) — compile-verified, matrix operand order checked by reading, not run on hardware
//
#include <hip/hip_runtime.h>
#include <hip/hip_bf16.h>

// ---------------------------------------------------------------------------
// MoE FFN for MI455X (gfx1250, wave32, WMMA bf16 16x16x32, f32 accum)
// Software-pipelined, double-buffered LDS GEMMs.
// ---------------------------------------------------------------------------

typedef __attribute__((ext_vector_type(16))) __bf16 v16bf;
typedef __attribute__((ext_vector_type(8)))  __bf16 v8bf;
typedef __attribute__((ext_vector_type(8)))  float  v8f;

static constexpr int kE    = 8;
static constexpr int kTOPK = 2;
static constexpr int kH    = 1024;
static constexpr int kF    = 4096;
static constexpr int kCAP  = 2048;
static constexpr int kN    = 8192;     // token copies

static constexpr int TM  = 128;        // GEMM tile M
static constexpr int TN  = 128;        // GEMM tile N
static constexpr int TK  = 64;         // GEMM tile K (2 wmma k-steps)
static constexpr int LDA = TK + 8;     // LDS row pitch (ushort) -> 144B, 16B aligned
static constexpr int BUF = 2 * TM * LDA;   // one A+B buffer, in ushorts (36 KB)

// ---- helpers --------------------------------------------------------------

__device__ __forceinline__ unsigned short f2bf(float f) {
  unsigned int u = __float_as_uint(f);
  unsigned int r = u + 0x7FFFu + ((u >> 16) & 1u);   // round-to-nearest-even
  return (unsigned short)(r >> 16);
}

__device__ __forceinline__ float gelu_exact(float v) {
  return 0.5f * v * (1.0f + erff(v * 0.70710678118654752f));
}

// Lane fragment from a K-major LDS row: elems 0-7 = K[half*8..), 8-15 = K[16+half*8..)
__device__ __forceinline__ v16bf frag_ld(const unsigned short* p, int half) {
  v8bf lo = *(const v8bf*)(p + half * 8);
  v8bf hi = *(const v8bf*)(p + 16 + half * 8);
  v16bf r;
#pragma unroll
  for (int j = 0; j < 8; ++j) { r[j] = lo[j]; r[j + 8] = hi[j]; }
  return r;
}

// ---- kernel 0: zero output (atomics accumulate into it) -------------------

__global__ __launch_bounds__(256) void zero_out_kernel(float* __restrict__ out) {
  int i = (blockIdx.x * 256 + threadIdx.x) * 4;
  *(float4*)(out + i) = make_float4(0.f, 0.f, 0.f, 0.f);
}

// ---- kernel 1: stable counting-sort routing (matches jnp.argsort) ---------

__global__ __launch_bounds__(256) void route_kernel(
    const int* __restrict__ experts,
    int* __restrict__ slot,       // [N]   e*CAP+rank, or -1 if dropped
    int* __restrict__ tok_slot,   // [E*CAP] slot -> token copy index
    int* __restrict__ counts) {   // [E]
  __shared__ int lcnt[256][kE];
  const int t = threadIdx.x;
  const int base = t * (kN / 256);           // 32 tokens per thread

  int c[kE];
#pragma unroll
  for (int e = 0; e < kE; ++e) c[e] = 0;
  for (int k = 0; k < kN / 256; ++k) c[experts[base + k]]++;
#pragma unroll
  for (int e = 0; e < kE; ++e) lcnt[t][e] = c[e];
  __syncthreads();

  if (t < kE) {                              // serial exclusive scan per expert
    int run = 0;
    for (int q = 0; q < 256; ++q) { int v = lcnt[q][t]; lcnt[q][t] = run; run += v; }
    counts[t] = run;
  }
  __syncthreads();

  int b[kE];
#pragma unroll
  for (int e = 0; e < kE; ++e) b[e] = lcnt[t][e];
  for (int k = 0; k < kN / 256; ++k) {
    int i = base + k;
    int e = experts[i];
    int r = b[e]++;
    if (r < kCAP) { int s = e * kCAP + r; slot[i] = s; tok_slot[s] = i; }
    else          { slot[i] = -1; }
  }
}

// ---- kernel 2: gather + f32->bf16 pack ------------------------------------

__global__ __launch_bounds__(256) void gather_kernel(
    const float* __restrict__ x, const int* __restrict__ slot,
    unsigned short* __restrict__ Xp) {
  const int i = blockIdx.x;
  const int s = slot[i];
  if (s < 0) return;
  const int t = threadIdx.x;
  float4 a = *(const float4*)(x + (size_t)i * kH + t * 4);
  *(ushort4*)(Xp + (size_t)s * kH + t * 4) =
      make_ushort4(f2bf(a.x), f2bf(a.y), f2bf(a.z), f2bf(a.w));
}

// ---- kernel 3: weight f32->bf16 conversion --------------------------------

__global__ __launch_bounds__(256) void wconv_kernel(
    const float* __restrict__ w1, const float* __restrict__ w2,
    unsigned short* __restrict__ W1b, unsigned short* __restrict__ W2b) {
  const float*    src = blockIdx.z ? w2  : w1;
  unsigned short* dst = blockIdx.z ? W2b : W1b;
  size_t base = ((size_t)blockIdx.x * 256 + threadIdx.x) * 8;
  float4 a = *(const float4*)(src + base);
  float4 b = *(const float4*)(src + base + 4);
  *(ushort4*)(dst + base)     = make_ushort4(f2bf(a.x), f2bf(a.y), f2bf(a.z), f2bf(a.w));
  *(ushort4*)(dst + base + 4) = make_ushort4(f2bf(b.x), f2bf(b.y), f2bf(b.z), f2bf(b.w));
}

// ---- kernel 4: fc1 = gelu(X @ W1^T) -> Hbuf (bf16) ------------------------
// grid: (F/TN, CAP/TM, E), 256 threads = 8 waves; wave -> 64x32 (4x2 wmma tiles)

__global__ __launch_bounds__(256) void fc1_kernel(
    const unsigned short* __restrict__ Xp,    // [E*CAP, H] bf16
    const unsigned short* __restrict__ W1b,   // [E, F, H]  bf16
    const int* __restrict__ counts,
    unsigned short* __restrict__ Hb) {        // [E*CAP, F] bf16
  const int e  = blockIdx.z;
  const int mt = blockIdx.y;
  const int nt = blockIdx.x;
  const int cnt = counts[e];
  if (mt * TM >= cnt) return;

  __shared__ __align__(16) unsigned short smem[2 * BUF];   // 72 KB, double buffered

  const int tid  = threadIdx.x;
  const int lane = tid & 31;
  const int wave = tid >> 5;
  const int wm   = wave & 1;        // 2 waves over M (64 rows each)
  const int wn   = wave >> 1;       // 4 waves over N (32 cols each)
  const int half = lane >> 4;
  const int l16  = lane & 15;

  v8f acc[4][2] = {};

  const unsigned short* Ag = Xp  + (size_t)(e * kCAP + mt * TM) * kH;
  const unsigned short* Bg = W1b + (size_t)e * kF * kH + (size_t)(nt * TN) * kH;

  const int lr = tid >> 1;          // 0..127: tile row to fetch
  const int lc = (tid & 1) * 32;    // K sub-chunk base
  const unsigned short* gA = Ag + (size_t)lr * kH + lc;
  const unsigned short* gB = Bg + (size_t)lr * kH + lc;
  const int sOff = lr * LDA + lc;   // staging destination within a buffer

  // prologue: stage tile 0 (distinct regs -> all loads in flight at once)
  uint4 ra[4], rb[4];
#pragma unroll
  for (int j = 0; j < 4; ++j) { ra[j] = *(const uint4*)(gA + j * 8);
                                rb[j] = *(const uint4*)(gB + j * 8); }
  {
    unsigned short* dA = smem;
    unsigned short* dB = smem + TM * LDA;
#pragma unroll
    for (int j = 0; j < 4; ++j) { *(uint4*)(dA + sOff + j * 8) = ra[j];
                                  *(uint4*)(dB + sOff + j * 8) = rb[j]; }
  }

  const int NK = kH / TK;
  for (int kt = 0; kt < NK; ++kt) {
    __syncthreads();
    const bool more = (kt + 1 < NK);
    if (more) {                      // issue next tile's global loads early
      const unsigned short* pa = gA + (kt + 1) * TK;
      const unsigned short* pb = gB + (kt + 1) * TK;
#pragma unroll
      for (int j = 0; j < 4; ++j) { ra[j] = *(const uint4*)(pa + j * 8);
                                    rb[j] = *(const uint4*)(pb + j * 8); }
    }

    const unsigned short* sA = smem + (kt & 1) * BUF;
    const unsigned short* sB = sA + TM * LDA;
#pragma unroll
    for (int ks = 0; ks < 2; ++ks) {
      const int kk = ks * 32;
      v16bf af[4], bfr[2];
#pragma unroll
      for (int i = 0; i < 4; ++i)
        af[i] = frag_ld(sA + (wm * 64 + i * 16 + l16) * LDA + kk, half);
#pragma unroll
      for (int j = 0; j < 2; ++j)
        bfr[j] = frag_ld(sB + (wn * 32 + j * 16 + l16) * LDA + kk, half);
#pragma unroll
      for (int i = 0; i < 4; ++i)
#pragma unroll
        for (int j = 0; j < 2; ++j)
          acc[i][j] = __builtin_amdgcn_wmma_f32_16x16x32_bf16(
              false, af[i], false, bfr[j], (short)0, acc[i][j], false, false);
    }

    if (more) {                      // fill the other buffer for next iter
      unsigned short* dA = smem + ((kt + 1) & 1) * BUF;
      unsigned short* dB = dA + TM * LDA;
#pragma unroll
      for (int j = 0; j < 4; ++j) { *(uint4*)(dA + sOff + j * 8) = ra[j];
                                    *(uint4*)(dB + sOff + j * 8) = rb[j]; }
    }
  }

  // epilogue: gelu -> bf16 -> repack via LDS -> coalesced b128 stores
  __syncthreads();
  unsigned short* sC = smem;        // [TM][TN] = 32 KB, reuse pipeline space
#pragma unroll
  for (int i = 0; i < 4; ++i)
#pragma unroll
    for (int j = 0; j < 2; ++j)
#pragma unroll
      for (int v = 0; v < 8; ++v) {
        int r = wm * 64 + i * 16 + half * 8 + v;   // C layout: VGPR v -> row
        int c = wn * 32 + j * 16 + l16;            // lane -> col
        sC[r * TN + c] = f2bf(gelu_exact(acc[i][j][v]));
      }
  __syncthreads();

  const int r  = tid >> 1;
  const int cb = (tid & 1) * 64;
  const int gm = mt * TM + r;
  if (gm < cnt) {
    size_t dst = (size_t)(e * kCAP + gm) * kF + nt * TN + cb;
#pragma unroll
    for (int j = 0; j < 8; ++j)
      *(uint4*)(Hb + dst + j * 8) = *(const uint4*)(sC + r * TN + cb + j * 8);
  }
}

// ---- kernel 5: fc2 = H @ W2^T, gate-scale, scatter-add into reduced out ---
// grid: (H/TN, CAP/TM, E)

__global__ __launch_bounds__(256) void fc2_kernel(
    const unsigned short* __restrict__ Hb,    // [E*CAP, F] bf16
    const unsigned short* __restrict__ W2b,   // [E, H, F]  bf16
    const int* __restrict__ counts,
    const int* __restrict__ tok_slot,         // [E*CAP]
    const float* __restrict__ gate,           // [N]
    const int* __restrict__ new_index,        // [N]
    float* __restrict__ out) {                // [N/TOPK, H]
  const int e  = blockIdx.z;
  const int mt = blockIdx.y;
  const int nt = blockIdx.x;
  const int cnt = counts[e];
  if (mt * TM >= cnt) return;

  __shared__ __align__(16) unsigned short smem[2 * BUF];

  const int tid  = threadIdx.x;
  const int lane = tid & 31;
  const int wave = tid >> 5;
  const int wm   = wave & 1;
  const int wn   = wave >> 1;
  const int half = lane >> 4;
  const int l16  = lane & 15;

  v8f acc[4][2] = {};

  const unsigned short* Ag = Hb  + (size_t)(e * kCAP + mt * TM) * kF;
  const unsigned short* Bg = W2b + (size_t)e * kH * kF + (size_t)(nt * TN) * kF;

  const int lr = tid >> 1;
  const int lc = (tid & 1) * 32;
  const unsigned short* gA = Ag + (size_t)lr * kF + lc;
  const unsigned short* gB = Bg + (size_t)lr * kF + lc;
  const int sOff = lr * LDA + lc;

  uint4 ra[4], rb[4];
#pragma unroll
  for (int j = 0; j < 4; ++j) { ra[j] = *(const uint4*)(gA + j * 8);
                                rb[j] = *(const uint4*)(gB + j * 8); }
  {
    unsigned short* dA = smem;
    unsigned short* dB = smem + TM * LDA;
#pragma unroll
    for (int j = 0; j < 4; ++j) { *(uint4*)(dA + sOff + j * 8) = ra[j];
                                  *(uint4*)(dB + sOff + j * 8) = rb[j]; }
  }

  const int NK = kF / TK;
  for (int kt = 0; kt < NK; ++kt) {
    __syncthreads();
    const bool more = (kt + 1 < NK);
    if (more) {
      const unsigned short* pa = gA + (kt + 1) * TK;
      const unsigned short* pb = gB + (kt + 1) * TK;
#pragma unroll
      for (int j = 0; j < 4; ++j) { ra[j] = *(const uint4*)(pa + j * 8);
                                    rb[j] = *(const uint4*)(pb + j * 8); }
    }

    const unsigned short* sA = smem + (kt & 1) * BUF;
    const unsigned short* sB = sA + TM * LDA;
#pragma unroll
    for (int ks = 0; ks < 2; ++ks) {
      const int kk = ks * 32;
      v16bf af[4], bfr[2];
#pragma unroll
      for (int i = 0; i < 4; ++i)
        af[i] = frag_ld(sA + (wm * 64 + i * 16 + l16) * LDA + kk, half);
#pragma unroll
      for (int j = 0; j < 2; ++j)
        bfr[j] = frag_ld(sB + (wn * 32 + j * 16 + l16) * LDA + kk, half);
#pragma unroll
      for (int i = 0; i < 4; ++i)
#pragma unroll
        for (int j = 0; j < 2; ++j)
          acc[i][j] = __builtin_amdgcn_wmma_f32_16x16x32_bf16(
              false, af[i], false, bfr[j], (short)0, acc[i][j], false, false);
    }

    if (more) {
      unsigned short* dA = smem + ((kt + 1) & 1) * BUF;
      unsigned short* dB = dA + TM * LDA;
#pragma unroll
      for (int j = 0; j < 4; ++j) { *(uint4*)(dA + sOff + j * 8) = ra[j];
                                    *(uint4*)(dB + sOff + j * 8) = rb[j]; }
    }
  }

  // stage per-row (out_row, gate) once per block into LDS
  __syncthreads();
  int*   sOrow = (int*)smem;
  float* sGate = (float*)(sOrow + TM);
  if (tid < TM) {
    const int gm = mt * TM + tid;
    int orow = 0; float g = 0.f;
    if (gm < cnt) {
      const int tok = tok_slot[e * kCAP + gm];
      g    = gate[tok];
      orow = new_index[tok] >> 1;          // /TOPK
    }
    sOrow[tid] = orow;
    sGate[tid] = g;
  }
  __syncthreads();

  // deterministic 2-way atomic combine: each out element gets <=2 float adds
#pragma unroll
  for (int i = 0; i < 4; ++i) {
#pragma unroll
    for (int v = 0; v < 8; ++v) {
      const int r  = wm * 64 + i * 16 + half * 8 + v;
      const int gm = mt * TM + r;
      if (gm < cnt) {
        const float g    = sGate[r];
        const int   orow = sOrow[r];
        float* ob = out + (size_t)orow * kH + nt * TN + wn * 32 + l16;
#pragma unroll
        for (int j = 0; j < 2; ++j)
          atomicAdd(ob + j * 16, g * acc[i][j][v]);
      }
    }
  }
}

// ---------------------------------------------------------------------------

extern "C" void kernel_launch(void* const* d_in, const int* in_sizes, int n_in,
                              void* d_out, int out_size, void* d_ws, size_t ws_size,
                              hipStream_t stream) {
  (void)in_sizes; (void)n_in; (void)out_size; (void)ws_size;

  const float* x         = (const float*)d_in[0];   // [N, H]
  const float* gate      = (const float*)d_in[1];   // [N]
  const int*   experts   = (const int*)d_in[2];     // [N]
  const int*   new_index = (const int*)d_in[3];     // [N]
  const float* w1        = (const float*)d_in[4];   // [E, F, H]
  const float* w2        = (const float*)d_in[5];   // [E, H, F]
  float*       out       = (float*)d_out;           // [N/TOPK, H]

  // workspace layout (all offsets 256B aligned)
  char* ws = (char*)d_ws;
  int* counts   = (int*)ws;                                    //      32 B
  int* slot     = (int*)(ws + 256);                            //  32 KiB
  int* tok_slot = (int*)(ws + 256 + 32768);                    //  64 KiB
  size_t off = 256 + 32768 + 65536;                            // = 98560
  unsigned short* Xp  = (unsigned short*)(ws + off);           //  32 MiB
  off += (size_t)kE * kCAP * kH * 2;
  unsigned short* W1b = (unsigned short*)(ws + off);           //  64 MiB
  off += (size_t)kE * kF * kH * 2;
  unsigned short* W2b = (unsigned short*)(ws + off);           //  64 MiB
  off += (size_t)kE * kH * kF * 2;
  unsigned short* Hb  = (unsigned short*)(ws + off);           // 128 MiB

  // 0) zero output (fc2 accumulates atomically)
  zero_out_kernel<<<(kN / kTOPK) * kH / (256 * 4), 256, 0, stream>>>(out);
  // 1) stable routing
  route_kernel<<<1, 256, 0, stream>>>(experts, slot, tok_slot, counts);
  // 2) pack activations to bf16
  gather_kernel<<<kN, 256, 0, stream>>>(x, slot, Xp);
  // 3) weights to bf16 (w1 and w2 via blockIdx.z)
  {
    const size_t elems = (size_t)kE * kF * kH;     // 33,554,432 per tensor
    dim3 g((unsigned)(elems / (256 * 8)), 1, 2);
    wconv_kernel<<<g, 256, 0, stream>>>(w1, w2, W1b, W2b);
  }
  // 4) fc1 + gelu
  fc1_kernel<<<dim3(kF / TN, kCAP / TM, kE), 256, 0, stream>>>(Xp, W1b, counts, Hb);
  // 5) fc2 + gate + scatter/top-2 combine
  fc2_kernel<<<dim3(kH / TN, kCAP / TM, kE), 256, 0, stream>>>(
      Hb, W2b, counts, tok_slot, gate, new_index, out);
}
